// LogNCDEFunc_9225589752315
// MI455X (gfx1250) — matrix operand order repaired and checked by reading, not compiled, not run on hardware
//
#include <hip/hip_runtime.h>
#include <hip/hip_bf16.h>
#include <math.h>

typedef _Float16 f16;
typedef _Float16 v16h __attribute__((ext_vector_type(16)));
typedef _Float16 v8h  __attribute__((ext_vector_type(8)));
typedef float    v8f  __attribute__((ext_vector_type(8)));

#define BB 16384
#define NN 64
#define HH 128

// ws layout (in f16 halves)
#define OFF_H  0
#define SZ_H   (BB * NN)            // 1,048,576 halves
#define OFF_W1 (OFF_H + SZ_H)
#define SZ_W1  (16 * 512)           // 2 k-tiles * 8 n-tiles
#define OFF_W2 (OFF_W1 + SZ_W1)
#define SZ_W2  (32 * 512)           // 4 k-tiles * 8 n-tiles
#define OFF_W3 (OFF_W2 + SZ_W2)
#define SZ_W3  (128 * 512)          // 4 k-tiles * 32 n-tiles

// ---------------- prep: f32 -> f16 convert + B-fragment swizzle ----------------
// B fragment layout (16-bit, 32x16): lane L holds N = L&15; halves e=0..15 map to
// K = 8*(L>>4) + (e&7) + 16*(e>>3). Stored so each (tile,lane)'s 16 halves are contiguous.
__global__ void prep_kernel(const float* __restrict__ h,
                            const float* __restrict__ W1,
                            const float* __restrict__ W2,
                            const float* __restrict__ W3,
                            f16* __restrict__ ws) {
  int idx = blockIdx.x * blockDim.x + threadIdx.x;
  if (idx < SZ_H) ws[OFF_H + idx] = (f16)h[idx];
  if (idx < SZ_W1) {
    int tile = idx >> 9, r = idx & 511, lane = r >> 4, e = r & 15;
    int nt = tile >> 1, kt = tile & 1;
    int K = kt * 32 + ((lane >> 4) << 3) + (e & 7) + ((e >> 3) << 4);
    int Nc = nt * 16 + (lane & 15);
    ws[OFF_W1 + idx] = (f16)W1[Nc * NN + K];
  }
  if (idx < SZ_W2) {
    int tile = idx >> 9, r = idx & 511, lane = r >> 4, e = r & 15;
    int nt = tile >> 2, kt = tile & 3;
    int K = kt * 32 + ((lane >> 4) << 3) + (e & 7) + ((e >> 3) << 4);
    int Nc = nt * 16 + (lane & 15);
    ws[OFF_W2 + idx] = (f16)W2[Nc * HH + K];
  }
  if (idx < SZ_W3) {
    int tile = idx >> 9, r = idx & 511, lane = r >> 4, e = r & 15;
    int nt = tile >> 2, kt = tile & 3;
    int K = kt * 32 + ((lane >> 4) << 3) + (e & 7) + ((e >> 3) << 4);
    int Nc = nt * 16 + (lane & 15);
    ws[OFF_W3 + idx] = (f16)W3[Nc * HH + K];
  }
}

// -------- fast branchless transcendentals (v_exp_f32 / v_log_f32 / v_rcp_f32) --
__device__ __forceinline__ float fsoftplus(float x) {
  // softplus(x) = max(x,0) + log(1 + exp(-|x|)); exp(-|x|) <= 1 -> no overflow
  return fmaxf(x, 0.f) + __logf(1.f + __expf(-fabsf(x)));
}
__device__ __forceinline__ float fsigmoid(float x) {
  return __builtin_amdgcn_rcpf(1.f + __expf(-x));
}
__device__ __forceinline__ float ftanh(float x) {
  float t = __expf(2.f * x);
  return 1.f - 2.f * __builtin_amdgcn_rcpf(t + 1.f);
}

// ---------------- main kernel helpers ----------------
__device__ __forceinline__ v8f wmma16(v16h a, v16h b, v8f c) {
  return __builtin_amdgcn_wmma_f32_16x16x32_f16(false, a, false, b, (short)0, c,
                                                false, false);
}

// A fragment (16-bit 16x32): lane L holds row M=L&15; halves 0..7 = cols
// [base+8*hi, +8), halves 8..15 = cols [base+16+8*hi, +8).
__device__ __forceinline__ v16h fragA(const f16* rowPtr, int colBase, int hi) {
  v16h a;
  ((v8h*)&a)[0] = *(const v8h*)(rowPtr + colBase + 8 * hi);
  ((v8h*)&a)[1] = *(const v8h*)(rowPtr + colBase + 16 + 8 * hi);
  return a;
}

__device__ __forceinline__ v16h fragB(const f16* __restrict__ wsB, int tile, int lane) {
  const v8h* p = (const v8h*)(wsB + (((size_t)tile << 5) + lane) * 16);
  v16h b;
  ((v8h*)&b)[0] = p[0];
  ((v8h*)&b)[1] = p[1];
  return b;
}

// ---------------- main kernel: one wave handles 16 batch rows ----------------
__global__ void __launch_bounds__(32, 4) logncde_kernel(
    const f16* __restrict__ ws, const float* __restrict__ sig,
    const float* __restrict__ b1, const float* __restrict__ b2,
    const float* __restrict__ b3, float* __restrict__ out) {
  // zu1/zu2: hold z1/z2 (activations, f16) during the primal pass, then are
  // reused as u1/u2 (JVP intermediates) during the direction passes.
  __shared__ __align__(16) f16 zu1[16 * 128];
  __shared__ __align__(16) f16 zu2[16 * 128];
  __shared__ __align__(16) f16 s1s[16 * 128];   // sigmoid(a1) masks
  __shared__ __align__(16) f16 s2s[16 * 128];   // sigmoid(a2) masks
  __shared__ __align__(16) f16 Vs[16 * 512];    // tanh(a3) = fields V
  __shared__ float sigs[16 * 36];

  const int l = threadIdx.x;  // 0..31
  const int ln = l & 15;      // M row within tile (for A) / N col (for C,D)
  const int hi = l >> 4;
  const int m0 = blockIdx.x * 16;

  const f16* hF = ws + OFF_H;
  const f16* wB1 = ws + OFF_W1;
  const f16* wB2 = ws + OFF_W2;
  const f16* wB3 = ws + OFF_W3;

  for (int idx = l; idx < 16 * 36; idx += 32) sigs[idx] = sig[(size_t)m0 * 36 + idx];

  // ---- primal layer 1: h(16x64) @ W1^T -> z1(16x128), s1 ----
  {
    const f16* hrow = hF + (size_t)(m0 + ln) * 64;
    v16h a0 = fragA(hrow, 0, hi);
    v16h a1 = fragA(hrow, 32, hi);
    for (int nt = 0; nt < 8; ++nt) {
      v8f acc = {};
      acc = wmma16(a0, fragB(wB1, nt * 2 + 0, l), acc);
      acc = wmma16(a1, fragB(wB1, nt * 2 + 1, l), acc);
      int col = nt * 16 + ln;
      float bias = b1[col];
      for (int r = 0; r < 8; ++r) {
        float x = acc[r] + bias;
        zu1[(r + 8 * hi) * 128 + col] = (f16)fsoftplus(x);
        s1s[(r + 8 * hi) * 128 + col] = (f16)fsigmoid(x);
      }
    }
  }
  __syncthreads();
  // ---- primal layer 2: z1 @ W2^T -> z2(16x128), s2 ----
  {
    v16h a[4];
    for (int kt = 0; kt < 4; ++kt) a[kt] = fragA(&zu1[ln * 128], kt * 32, hi);
    for (int nt = 0; nt < 8; ++nt) {
      v8f acc = {};
      for (int kt = 0; kt < 4; ++kt) acc = wmma16(a[kt], fragB(wB2, nt * 4 + kt, l), acc);
      int col = nt * 16 + ln;
      float bias = b2[col];
      for (int r = 0; r < 8; ++r) {
        float x = acc[r] + bias;
        zu2[(r + 8 * hi) * 128 + col] = (f16)fsoftplus(x);
        s2s[(r + 8 * hi) * 128 + col] = (f16)fsigmoid(x);
      }
    }
  }
  __syncthreads();
  // ---- primal layer 3: z2 @ W3^T -> V(16x512) = tanh ----
  {
    v16h a[4];
    for (int kt = 0; kt < 4; ++kt) a[kt] = fragA(&zu2[ln * 128], kt * 32, hi);
    for (int nt = 0; nt < 32; ++nt) {
      v8f acc = {};
      for (int kt = 0; kt < 4; ++kt) acc = wmma16(a[kt], fragB(wB3, nt * 4 + kt, l), acc);
      int col = nt * 16 + ln;
      float bias = b3[col];
      for (int r = 0; r < 8; ++r)
        Vs[(r + 8 * hi) * 512 + col] = (f16)ftanh(acc[r] + bias);
    }
  }
  __syncthreads();

  // ---- level-1 contraction: out = sig[:8] @ V ----
  v8f outAcc[4];
  for (int t = 0; t < 4; ++t) {
    v8f o = {};
    for (int r = 0; r < 8; ++r) {
      int row = r + 8 * hi;
      float s = 0.f;
      for (int i = 0; i < 8; ++i)
        s += sigs[row * 36 + i] * (float)Vs[row * 512 + i * 64 + t * 16 + ln];
      o[r] = s;
    }
    outAcc[t] = o;
  }

  // ---- JVP passes: direction i = V_i; accumulate brackets on the fly ----
  for (int i = 0; i < 8; ++i) {
    __syncthreads();  // zu1/zu2 reuse across iterations
    {  // layer-1 JVP: u1 = (T @ W1^T) * s1
      v16h t0 = fragA(&Vs[ln * 512 + i * 64], 0, hi);
      v16h t1 = fragA(&Vs[ln * 512 + i * 64], 32, hi);
      for (int nt = 0; nt < 8; ++nt) {
        v8f acc = {};
        acc = wmma16(t0, fragB(wB1, nt * 2 + 0, l), acc);
        acc = wmma16(t1, fragB(wB1, nt * 2 + 1, l), acc);
        int col = nt * 16 + ln;
        for (int r = 0; r < 8; ++r) {
          int row = r + 8 * hi;
          float s1 = (float)s1s[row * 128 + col];
          zu1[row * 128 + col] = (f16)(acc[r] * s1);
        }
      }
    }
    __syncthreads();
    {  // layer-2 JVP: u2 = (u1 @ W2^T) * s2
      v16h a[4];
      for (int kt = 0; kt < 4; ++kt) a[kt] = fragA(&zu1[ln * 128], kt * 32, hi);
      for (int nt = 0; nt < 8; ++nt) {
        v8f acc = {};
        for (int kt = 0; kt < 4; ++kt) acc = wmma16(a[kt], fragB(wB2, nt * 4 + kt, l), acc);
        int col = nt * 16 + ln;
        for (int r = 0; r < 8; ++r) {
          int row = r + 8 * hi;
          float s2 = (float)s2s[row * 128 + col];
          zu2[row * 128 + col] = (f16)(acc[r] * s2);
        }
      }
    }
    __syncthreads();
    {  // layer-3 JVP, chunk j -> J_{V_j} V_i, fused bracket accumulation
      v16h a[4];
      for (int kt = 0; kt < 4; ++kt) a[kt] = fragA(&zu2[ln * 128], kt * 32, hi);
      for (int j = 0; j < 8; ++j) {
        if (j == i) continue;  // [V_i, V_i] = 0
        int p = (i < j) ? (7 * i - (i * (i - 1)) / 2 + (j - i - 1))
                        : (7 * j - (j * (j - 1)) / 2 + (i - j - 1));
        float sgn = (i < j) ? 1.f : -1.f;
        float cf[8];
        for (int r = 0; r < 8; ++r)
          cf[r] = sgn * sigs[(r + 8 * hi) * 36 + 8 + p];
        for (int t = 0; t < 4; ++t) {
          int nt = j * 4 + t;
          v8f acc = {};
          for (int kt = 0; kt < 4; ++kt)
            acc = wmma16(a[kt], fragB(wB3, nt * 4 + kt, l), acc);
          int col = nt * 16 + ln;
          for (int r = 0; r < 8; ++r) {
            int row = r + 8 * hi;
            float v = (float)Vs[row * 512 + col];
            float g = 1.f - v * v;  // tanh'
            outAcc[t][r] += cf[r] * (g * acc[r]);
          }
        }
      }
    }
  }

  // ---- store (B,64) f32 ----
  for (int t = 0; t < 4; ++t)
    for (int r = 0; r < 8; ++r)
      out[(size_t)(m0 + r + 8 * hi) * 64 + t * 16 + ln] = outAcc[t][r];
}

extern "C" void kernel_launch(void* const* d_in, const int* in_sizes, int n_in,
                              void* d_out, int out_size, void* d_ws, size_t ws_size,
                              hipStream_t stream) {
  const float* h = (const float*)d_in[0];
  const float* sg = (const float*)d_in[1];
  const float* W1 = (const float*)d_in[2];
  const float* b1 = (const float*)d_in[3];
  const float* W2 = (const float*)d_in[4];
  const float* b2 = (const float*)d_in[5];
  const float* W3 = (const float*)d_in[6];
  const float* b3 = (const float*)d_in[7];
  f16* ws = (f16*)d_ws;

  prep_kernel<<<(SZ_H + 255) / 256, 256, 0, stream>>>(h, W1, W2, W3, ws);
  logncde_kernel<<<BB / 16, 32, 0, stream>>>(ws, sg, b1, b2, b3, (float*)d_out);
}
